// HeroEmbedding_6511170421580
// MI455X (gfx1250) — matrix operand order repaired and checked by reading, not compiled, not run on hardware
//
#include <hip/hip_runtime.h>

typedef __attribute__((ext_vector_type(16))) __bf16 v16bf;
typedef __attribute__((ext_vector_type(8)))  float  v8f;

#define H     121          // hero count
#define RNK   9            // rank size
#define V     128          // output vector width
#define NB    65536        // batch
#define BIASW 193          // used columns of W_bias (9*8+120+1)
#define KTOT  352          // 121 + 9 + 193 = 323 -> pad to 11*32
#define LDSK  360          // padded LDS row stride (16B aligned, bank spread)
#define WAVES 4
#define TILES 8            // 16-row tiles per wave -> 512 rows per block

union Frag { v16bf v; uint4 q[2]; };

__device__ __forceinline__ unsigned short f2bf(float f) {
    unsigned u = __builtin_bit_cast(unsigned, f);
    u += 0x7FFFu + ((u >> 16) & 1u);      // round-to-nearest-even
    return (unsigned short)(u >> 16);
}

extern "C" __global__ __launch_bounds__(WAVES * 32)
void hero_embed_wmma(const float* __restrict__ heroes,
                     const float* __restrict__ rank,
                     const float* __restrict__ Wb,
                     const float* __restrict__ Ws,
                     const float* __restrict__ Wx,
                     float* __restrict__ out)
{
    extern __shared__ unsigned short smem[];
    unsigned short* Wc = smem;                 // [V][LDSK] combined bf16 weights
    unsigned short* At = smem + V * LDSK;      // [WAVES][16][LDSK] shifted A tiles

    const int tid  = threadIdx.x;
    const int lane = tid & 31;
    const int wave = tid >> 5;

    // ---- stage combined weight matrix into LDS (one row per thread) ----
    {
        const int v = tid;                                 // 0..127
        const float* wbrow = Wb + v * H;
        const float* wsrow = Ws + v * RNK;
        const float* wxrow = Wx + v * (RNK * H);
        unsigned short* dst = Wc + v * LDSK;
        for (int k = 0; k < LDSK; ++k) {
            float w;
            if (k < H)                 w = wbrow[k];
            else if (k < H + RNK)      w = wsrow[k - H];
            else if (k < H + RNK + BIASW) w = wxrow[k - (H + RNK)];
            else                       w = 0.f;
            dst[k] = f2bf(w);
        }
    }
    __syncthreads();

    const int m  = lane & 15;    // column / row-in-tile index
    const int hi = lane >> 4;    // lane half

    for (int it = 0; it < TILES; ++it) {
        const int b0 = blockIdx.x * (WAVES * 16 * TILES) + it * (WAVES * 16) + wave * 16;

        // ---- build shifted bf16 A tile in LDS: 2 lanes per row ----
        {
            const int row = lane >> 1;
            const int b   = b0 + row;
            const float* hrow = heroes + (size_t)b * H;
            const float* rrow = rank   + (size_t)b * RNK;
            int rid = 0; float best = rrow[0];
            #pragma unroll
            for (int i = 1; i < RNK; ++i) { float x = rrow[i]; if (x > best) { best = x; rid = i; } }
            const int shift = H + RNK + 9 * rid;  // k where shifted heroes start
            unsigned short* arow = At + (wave * 16 + row) * LDSK;
            const int k0 = (lane & 1) * (LDSK / 2);
            for (int k = k0; k < k0 + LDSK / 2; ++k) {
                float x;
                if (k < H)            x = hrow[k];
                else if (k < H + RNK) x = rrow[k - H];
                else {
                    int j = k - shift;                      // pad region auto-zeros
                    x = (j >= 0 && j < H) ? hrow[j] : 0.f;
                }
                arow[k] = f2bf(x);
            }
            if (it + 1 < TILES)                              // warm next tile's rows
                __builtin_prefetch(heroes + (size_t)(b + WAVES * 16) * H, 0, 0);
        }
        __syncthreads();

        // ---- A fragments (16-bit A 16x32 layout: lanes<16 hold K 0..7,16..23) ----
        Frag afr[KTOT / 32];
        {
            const unsigned short* ar = At + (wave * 16 + m) * LDSK;
            #pragma unroll
            for (int s = 0; s < KTOT / 32; ++s) {
                const unsigned short* p = ar + s * 32 + hi * 8;
                afr[s].q[0] = *(const uint4*)(p);
                afr[s].q[1] = *(const uint4*)(p + 16);
            }
        }

        // ---- 8 V-tiles of 16 columns, stream B fragments from LDS ----
        for (int vt = 0; vt < 8; ++vt) {
            v8f acc = {};
            const unsigned short* br = Wc + (vt * 16 + m) * LDSK;
            #pragma unroll
            for (int s = 0; s < KTOT / 32; ++s) {
                Frag bfr;   // B 32x16 layout: lanes<16 hold K 0..7,16..23
                const unsigned short* p = br + s * 32 + hi * 8;
                bfr.q[0] = *(const uint4*)(p);
                bfr.q[1] = *(const uint4*)(p + 16);
                acc = __builtin_amdgcn_wmma_f32_16x16x32_bf16(
                          false, afr[s].v, false, bfr.v, (short)0, acc, false, false);
            }
            // C layout: VGPR i -> row hi*8+i, col m
            float* op = out + (size_t)(b0 + hi * 8) * V + vt * 16 + m;
            #pragma unroll
            for (int i = 0; i < 8; ++i) op[(size_t)i * V] = acc[i];
        }
        __syncthreads();
    }
}

extern "C" void kernel_launch(void* const* d_in, const int* in_sizes, int n_in,
                              void* d_out, int out_size, void* d_ws, size_t ws_size,
                              hipStream_t stream) {
    const float* heroes = (const float*)d_in[0];
    const float* rank   = (const float*)d_in[1];
    const float* Wb     = (const float*)d_in[2];
    const float* Ws     = (const float*)d_in[3];
    const float* Wx     = (const float*)d_in[4];
    float* out = (float*)d_out;

    const size_t shmem = (size_t)(V + WAVES * 16) * LDSK * sizeof(unsigned short); // 138240 B
    hipFuncSetAttribute((const void*)hero_embed_wmma,
                        hipFuncAttributeMaxDynamicSharedMemorySize, (int)shmem);

    const int rows_per_block = WAVES * 16 * TILES;   // 512
    hipLaunchKernelGGL(hero_embed_wmma, dim3(NB / rows_per_block), dim3(WAVES * 32),
                       shmem, stream, heroes, rank, Wb, Ws, Wx, out);
}
